// HypergraphNodeBlock_28286654612011
// MI455X (gfx1250) — matrix operand order repaired and checked by reading, not compiled, not run on hardware
//
#include <hip/hip_runtime.h>

// ---------------- problem constants ----------------
#define NTOT   (64 * 4096)     // 262144 nodes
#define EEDG   524288          // hyperedges
#define DN     128
#define DG     32
#define DE1    64
#define DE2    32
#define DIN    256             // DN+DG+DE1+DE2
#define H1     256
#define H2     128
#define MTILE  64              // nodes per block (4 x 16-row WMMA tiles)

typedef __attribute__((ext_vector_type(16))) __bf16 bf16x16;
typedef __attribute__((ext_vector_type(8)))  __bf16 bf16x8;
typedef __attribute__((ext_vector_type(8)))  float  f32x8;

// ---------------- workspace zero ----------------
__global__ void zero_f32_kernel(float* __restrict__ p, long n) {
    long i = (long)blockIdx.x * blockDim.x + threadIdx.x;
    long stride = (long)gridDim.x * blockDim.x;
    for (; i < n; i += stride) p[i] = 0.0f;
}

// ---------------- pack f32 weights [K=256, Ncols] into bf16 WMMA B-fragments ----
// Fragment layout (per 32x16 K-tile x N-tile):
//   lane 0..15  : N = nt*16 + lane,      elements e=0..7 -> K = kt*32 + e,     e=8..15 -> K = kt*32 + 8 + e
//   lane 16..31 : N = nt*16 + (lane-16), elements e=0..7 -> K = kt*32 + 8 + e, e=8..15 -> K = kt*32 + 16 + e
// Stored contiguously: frag[((nt*8 + kt)*32 + lane)*16 + e]
__global__ void pack_B_kernel(const float* __restrict__ W, __bf16* __restrict__ out,
                              int ncols, int total) {
    int t = blockIdx.x * blockDim.x + threadIdx.x;
    if (t >= total) return;
    int e    = t & 15;
    int lane = (t >> 4) & 31;
    int kt   = (t >> 9) & 7;
    int nt   = t >> 12;
    int n = nt * 16 + (lane & 15);
    int k = kt * 32 + ((lane < 16) ? 0 : 8) + (e & 7) + ((e >> 3) << 4);
    out[t] = (__bf16)W[k * ncols + n];
}

// ---------------- scatter: each hyperedge adds its 64-vec to 4 nodes ----------
__global__ void scatter_k4_kernel(const float* __restrict__ feat,
                                  const int* __restrict__ idx,
                                  float* __restrict__ agg) {
    int t = blockIdx.x * blockDim.x + threadIdx.x;   // over EEDG*64
    int e = t >> 6;
    int f = t & 63;
    float v = feat[t];
#pragma unroll
    for (int k = 0; k < 4; ++k) {
        int node = idx[e * 4 + k];
        atomicAdd(&agg[(long)node * DE1 + f], v);    // global_atomic_add_f32
    }
}

// ---------------- scatter: unsorted segment sum of 32-vecs -------------------
__global__ void scatter_seg_kernel(const float* __restrict__ feat,
                                   const int* __restrict__ seg,
                                   float* __restrict__ agg) {
    int t = blockIdx.x * blockDim.x + threadIdx.x;   // over EEDG*32
    int e = t >> 5;
    int f = t & 31;
    atomicAdd(&agg[(long)seg[e] * DE2 + f], feat[t]);
}

// ---------------- A-fragment loader from row-major bf16 LDS tile (stride 256) -
// 16-bit A 16x32 layout: lanes 0-15 (M=lane) hold K=0..7 then K=16..23;
// lanes 16-31 (M=lane-16) hold K=8..15 then K=24..31.  Two ds_load_b128.
__device__ __forceinline__ bf16x16 load_afrag(const __bf16* base, int lane, int kt) {
    int m  = lane & 15;
    int kb = kt * 32 + ((lane < 16) ? 0 : 8);
    bf16x8 lo = *(const bf16x8*)(base + m * 256 + kb);
    bf16x8 hi = *(const bf16x8*)(base + m * 256 + kb + 16);
    return __builtin_shufflevector(lo, hi, 0, 1, 2, 3, 4, 5, 6, 7,
                                           8, 9, 10, 11, 12, 13, 14, 15);
}

// ---------------- fused: compose x (bf16) -> GEMM1 -> GEMM2 -> LayerNorm -----
// Block = 256 threads = 8 waves, processes MTILE=64 nodes (4 WMMA row-tiles).
// Each B fragment is loaded once per kt and reused across the 4 A row-tiles.
__launch_bounds__(256)
__global__ void fused_mlp_ln_kernel(const float* __restrict__ nodes,
                                    const float* __restrict__ gfeat,
                                    const float* __restrict__ agg1,
                                    const float* __restrict__ agg2,
                                    const __bf16* __restrict__ W1b,
                                    const float* __restrict__ b1,
                                    const __bf16* __restrict__ W2b,
                                    const float* __restrict__ b2,
                                    const float* __restrict__ gamma,
                                    const float* __restrict__ beta,
                                    float* __restrict__ out) {
    __shared__ __align__(16) __bf16 Xs[MTILE * 256];   // 32 KB: x tile (bf16)
    __shared__ __align__(16) __bf16 Hs[MTILE * 256];   // 32 KB: hidden1 tile (bf16)
    // hidden2 f32 tile (32 KB) aliases Xs, which is dead after stage 2
    float* H2f = (float*)Xs;

    const int  tid  = threadIdx.x;
    const int  lane = tid & 31;
    const int  wave = tid >> 5;
    const long nodeBase = (long)blockIdx.x * MTILE;

    // prefetch this wave's weight-fragment streams (global_prefetch_b8)
    __builtin_prefetch(W1b + ((wave * 2 * 8) * 32 + lane) * 16, 0, 0);
    __builtin_prefetch(W2b + ((wave * 8) * 32 + lane) * 16, 0, 0);

    // ---- stage 1: compose concat([nodes, g_rep, agg1, agg2]) as bf16 in LDS
#pragma unroll
    for (int rr = 0; rr < 4; ++rr) {
        int  r    = rr * 16 + (tid >> 4);   // row 0..63
        int  c0   = (tid & 15) * 16;        // 16-col chunk, never straddles segments
        long node = nodeBase + r;
        const float* src;
        int off;
        if (c0 < DN)                 { src = nodes + node * DN;         off = c0; }
        else if (c0 < DN + DG)       { src = gfeat + (node >> 12) * DG; off = c0 - DN; }
        else if (c0 < DN + DG + DE1) { src = agg1 + node * DE1;         off = c0 - DN - DG; }
        else                         { src = agg2 + node * DE2;         off = c0 - DN - DG - DE1; }
#pragma unroll
        for (int j = 0; j < 16; j += 4) {
            float4 v = *(const float4*)(src + off + j);
            Xs[r * 256 + c0 + j + 0] = (__bf16)v.x;
            Xs[r * 256 + c0 + j + 1] = (__bf16)v.y;
            Xs[r * 256 + c0 + j + 2] = (__bf16)v.z;
            Xs[r * 256 + c0 + j + 3] = (__bf16)v.w;
        }
    }
    __syncthreads();

    // ---- stage 2: h1 = relu(x @ W1 + b1); wave w owns H1 cols [w*32, w*32+32)
    {
        f32x8 acc[4][2] = {};
        const int nt0 = wave * 2, nt1 = wave * 2 + 1;
#pragma unroll
        for (int kt = 0; kt < 8; ++kt) {
            bf16x16 bv0 = *(const bf16x16*)(W1b + ((nt0 * 8 + kt) * 32 + lane) * 16);
            bf16x16 bv1 = *(const bf16x16*)(W1b + ((nt1 * 8 + kt) * 32 + lane) * 16);
#pragma unroll
            for (int mt = 0; mt < 4; ++mt) {
                bf16x16 a = load_afrag(Xs + mt * 16 * 256, lane, kt);
                acc[mt][0] = __builtin_amdgcn_wmma_f32_16x16x32_bf16(
                    false, a, false, bv0, (short)0, acc[mt][0], false, false);
                acc[mt][1] = __builtin_amdgcn_wmma_f32_16x16x32_bf16(
                    false, a, false, bv1, (short)0, acc[mt][1], false, false);
            }
        }
        const int n0 = nt0 * 16 + (lane & 15);
        const int n1 = n0 + 16;
        const float bias0 = b1[n0];
        const float bias1 = b1[n1];
        const int mbase = (lane < 16) ? 0 : 8;   // C/D layout: lanes 16-31 hold M=8..15
#pragma unroll
        for (int mt = 0; mt < 4; ++mt) {
#pragma unroll
            for (int r = 0; r < 8; ++r) {
                float h0 = acc[mt][0][r] + bias0; h0 = h0 > 0.0f ? h0 : 0.0f;
                float h1 = acc[mt][1][r] + bias1; h1 = h1 > 0.0f ? h1 : 0.0f;
                Hs[(mt * 16 + mbase + r) * 256 + n0] = (__bf16)h0;
                Hs[(mt * 16 + mbase + r) * 256 + n1] = (__bf16)h1;
            }
        }
    }
    __syncthreads();   // Xs fully consumed; its LDS is reused as H2f below

    // ---- stage 3: h2 = relu(h1 @ W2 + b2); wave w owns H2 cols [w*16, w*16+16)
    {
        f32x8 acc[4] = {};
        const int nt = wave;
#pragma unroll
        for (int kt = 0; kt < 8; ++kt) {
            bf16x16 bv = *(const bf16x16*)(W2b + ((nt * 8 + kt) * 32 + lane) * 16);
#pragma unroll
            for (int mt = 0; mt < 4; ++mt) {
                bf16x16 a = load_afrag(Hs + mt * 16 * 256, lane, kt);
                acc[mt] = __builtin_amdgcn_wmma_f32_16x16x32_bf16(
                    false, a, false, bv, (short)0, acc[mt], false, false);
            }
        }
        const int n = nt * 16 + (lane & 15);
        const float bias = b2[n];
        const int mbase = (lane < 16) ? 0 : 8;
#pragma unroll
        for (int mt = 0; mt < 4; ++mt) {
#pragma unroll
            for (int r = 0; r < 8; ++r) {
                float h = acc[mt][r] + bias; h = h > 0.0f ? h : 0.0f;
                H2f[(mt * 16 + mbase + r) * 128 + n] = h;
            }
        }
    }
    __syncthreads();

    // ---- stage 4: LayerNorm over 128 features; 16 threads/row, 8 cols each
#pragma unroll
    for (int rr = 0; rr < 4; ++rr) {
        int r  = rr * 16 + (tid >> 4);   // row 0..63
        int c0 = (tid & 15) * 8;         // col base
        float v[8];
        float s = 0.0f, ss = 0.0f;
#pragma unroll
        for (int j = 0; j < 8; ++j) {
            v[j] = H2f[r * 128 + c0 + j];
            s += v[j];
            ss += v[j] * v[j];
        }
#pragma unroll
        for (int off = 1; off < 16; off <<= 1) {
            s  += __shfl_xor(s,  off, 16);
            ss += __shfl_xor(ss, off, 16);
        }
        const float mu   = s * (1.0f / 128.0f);
        const float var  = ss * (1.0f / 128.0f) - mu * mu;
        const float rstd = rsqrtf(var + 1e-3f);
        const long  node = nodeBase + r;
#pragma unroll
        for (int j = 0; j < 8; ++j) {
            int cc = c0 + j;
            out[node * 128 + cc] = (v[j] - mu) * rstd * gamma[cc] + beta[cc];
        }
    }
}

// ---------------- launch ----------------
extern "C" void kernel_launch(void* const* d_in, const int* in_sizes, int n_in,
                              void* d_out, int out_size, void* d_ws, size_t ws_size,
                              hipStream_t stream) {
    const float* nodes    = (const float*)d_in[0];
    const float* gfeat    = (const float*)d_in[1];
    // d_in[2] = n_node (constant 4096 per graph; node->graph is node>>12)
    const float* he1_feat = (const float*)d_in[3];
    const int*   he1_idx  = (const int*)d_in[4];
    const float* he2_feat = (const float*)d_in[5];
    const int*   he2_seg  = (const int*)d_in[6];
    const float* W1       = (const float*)d_in[7];
    const float* b1       = (const float*)d_in[8];
    const float* W2       = (const float*)d_in[9];
    const float* b2       = (const float*)d_in[10];
    const float* gamma    = (const float*)d_in[11];
    const float* beta     = (const float*)d_in[12];

    char* ws = (char*)d_ws;
    __bf16* W1b = (__bf16*)ws;                                   // 256*256*2   = 128 KB
    __bf16* W2b = (__bf16*)(ws + 256 * 256 * 2);                 // 256*128*2   =  64 KB
    float*  agg1 = (float*)(ws + 256 * 256 * 2 + 256 * 128 * 2); // N*64*4
    float*  agg2 = agg1 + (long)NTOT * DE1;                      // N*32*4

    // 1) zero the aggregation buffers (agg1 and agg2 are contiguous)
    zero_f32_kernel<<<4096, 256, 0, stream>>>(agg1, (long)NTOT * (DE1 + DE2));

    // 2) repack weights into bf16 WMMA B-fragment layout
    pack_B_kernel<<<(256 * H1) / 256, 256, 0, stream>>>(W1, W1b, H1, 256 * H1);
    pack_B_kernel<<<(256 * H2) / 256, 256, 0, stream>>>(W2, W2b, H2, 256 * H2);

    // 3) scatter-sum hyperedge features onto nodes (f32 global atomics)
    scatter_k4_kernel<<<(EEDG * DE1) / 256, 256, 0, stream>>>(he1_feat, he1_idx, agg1);
    scatter_seg_kernel<<<(EEDG * DE2) / 256, 256, 0, stream>>>(he2_feat, he2_seg, agg2);

    // 4) fused compose + MLP (bf16 WMMA, f32 accumulate) + LayerNorm
    fused_mlp_ln_kernel<<<NTOT / MTILE, 256, 0, stream>>>(
        nodes, gfeat, agg1, agg2, W1b, b1, W2b, b2, gamma, beta, (float*)d_out);
}
